// TransformerBlock_19842748907606
// MI455X (gfx1250) — matrix operand order repaired and checked by reading, not compiled, not run on hardware
//
#include <hip/hip_runtime.h>
#include <math.h>

// ---------------- problem dims ----------------
#define Ndim 8
#define Sdim 1024
#define Edim 768
#define Hdim 12
#define Ddim 64
#define FFdim 3072
#define Mdim (Ndim * Sdim)   // 8192 tokens

typedef _Float16 f16;
typedef __attribute__((ext_vector_type(16))) _Float16 v16h;
typedef __attribute__((ext_vector_type(8)))  _Float16 v8h;
typedef __attribute__((ext_vector_type(8)))  float    v8f;

#define WMMA_F16(a, b, c) \
  __builtin_amdgcn_wmma_f32_16x16x32_f16(false, (a), false, (b), (short)0, (c), false, false)

// GEMM epilogue modes
#define MODE_QKV  0
#define MODE_RES  1
#define MODE_GELU 2

// ---------------------------------------------------------------------------
// Fragment loaders (wave32, 16x16x32 f16 WMMA layouts per CDNA5 ISA)
// A: 16x32 (MxK), row-major source [*, lda]. Lane l: row = l&15;
//    elems 0..7  -> K = k0 + (l>>4)*8 + 0..7
//    elems 8..15 -> K = k0 + 16 + (l>>4)*8 + 0..7
// ---------------------------------------------------------------------------
__device__ __forceinline__ v16h load_a_frag(const f16* __restrict__ A, int lda,
                                            int row0, int k0) {
  int lane = threadIdx.x & 31;
  int m  = lane & 15;
  int hl = lane >> 4;
  const f16* p = A + (size_t)(row0 + m) * lda + k0 + hl * 8;
  v8h lo = *(const v8h*)p;
  v8h hi = *(const v8h*)(p + 16);
  v16h r;
#pragma unroll
  for (int i = 0; i < 8; ++i) { r[i] = lo[i]; r[i + 8] = hi[i]; }
  return r;
}

// B: 32x16 (KxN), loaded from BT stored row-major as [N, ldk].
// Lane l: col n = n0 + (l&15); elems 0..7 -> K = k0 + (l>>4)*8 + e, elems 8..15 -> K = k0 + 16 + (l>>4)*8 + (e-8).
__device__ __forceinline__ v16h load_bt_frag(const f16* __restrict__ BT, int ldk,
                                             int n0, int k0) {
  int lane = threadIdx.x & 31;
  int n  = n0 + (lane & 15);
  int kk = k0 + (lane >> 4) * 8;
  const f16* p = BT + (size_t)n * ldk + kk;
  v8h lo = *(const v8h*)p;
  v8h hi = *(const v8h*)(p + 16);
  v16h r;
#pragma unroll
  for (int i = 0; i < 8; ++i) { r[i] = lo[i]; r[i + 8] = hi[i]; }
  return r;
}

// ---------------------------------------------------------------------------
// Weight prep: f32 [K,N] row-major  ->  f16 transposed [N,K]
// ---------------------------------------------------------------------------
__global__ void convert_transpose_kernel(const float* __restrict__ src,
                                         f16* __restrict__ dst, int K, int N) {
  size_t idx = (size_t)blockIdx.x * 256 + threadIdx.x;
  size_t total = (size_t)K * N;
  if (idx < total) {
    int k = (int)(idx / N);
    int n = (int)(idx - (size_t)k * N);
    dst[(size_t)n * K + k] = (f16)src[idx];
  }
}

// ---------------------------------------------------------------------------
// LayerNorm: f32 [M, E] -> f16 [M, E]; one block (256 thr) per row.
// ---------------------------------------------------------------------------
__global__ void layernorm_f16_kernel(const float* __restrict__ x,
                                     const float* __restrict__ g,
                                     const float* __restrict__ b,
                                     f16* __restrict__ out) {
  int row = blockIdx.x;
  int tid = threadIdx.x;
  const float* xr = x + (size_t)row * Edim;

  float s = 0.f, ss = 0.f;
  for (int i = tid; i < Edim; i += 256) { float v = xr[i]; s += v; ss += v * v; }
#pragma unroll
  for (int off = 16; off > 0; off >>= 1) {
    s  += __shfl_down(s,  off);
    ss += __shfl_down(ss, off);
  }
  __shared__ float as[8], ass[8];
  if ((tid & 31) == 0) { as[tid >> 5] = s; ass[tid >> 5] = ss; }
  __syncthreads();
  float S = 0.f, SS = 0.f;
#pragma unroll
  for (int w = 0; w < 8; ++w) { S += as[w]; SS += ass[w]; }
  float mu   = S * (1.0f / Edim);
  float var  = SS * (1.0f / Edim) - mu * mu;
  float rstd = rsqrtf(var + 1e-5f);

  f16* orow = out + (size_t)row * Edim;
  for (int i = tid; i < Edim; i += 256) {
    float v = (xr[i] - mu) * rstd * g[i] + b[i];
    orow[i] = (f16)v;
  }
}

// ---------------------------------------------------------------------------
// Generic WMMA GEMM: C[M,Nc] = A[M,K] @ B[K,Nc] + bias, fused epilogue.
// Block = 256 thr = 8 waves in a 2x4 grid; each wave owns a 32(M)x64(N) tile
// (2 A frags x 4 B frags -> 8 WMMA per K-step, full cross reuse).
// Software pipelined with a PEELED final step: the steady-state body issues
// the k+32 fragment loads unconditionally, then runs 8 WMMAs on the k
// fragments, so load latency hides under the matrix burst. launch_bounds
// (256, 1) relaxes the occupancy target so ~190 live VGPRs fit w/o spills.
// ---------------------------------------------------------------------------
__global__ void __launch_bounds__(256, 1)
gemm_wmma_kernel(const f16* __restrict__ A,
                 const f16* __restrict__ BT,
                 const float* __restrict__ bias,
                 int M, int K, int Nc, int mode,
                 float* __restrict__ out_f32,
                 const float* __restrict__ res_in,
                 f16* __restrict__ out_f16,
                 f16* __restrict__ qb,
                 f16* __restrict__ kb,
                 f16* __restrict__ vtb) {
  int wave = threadIdx.x >> 5;        // 0..7
  int wrow = wave >> 2;               // 0..1
  int wcol = wave & 3;                // 0..3
  int row0 = blockIdx.y * 64 + wrow * 32;
  int col0 = blockIdx.x * 256 + wcol * 64;

  v8f acc[2][4] = {};

  // prologue: fragments for k = 0
  v16h a0 = load_a_frag(A, K, row0,      0);
  v16h a1 = load_a_frag(A, K, row0 + 16, 0);
  v16h b0 = load_bt_frag(BT, K, col0,      0);
  v16h b1 = load_bt_frag(BT, K, col0 + 16, 0);
  v16h b2 = load_bt_frag(BT, K, col0 + 32, 0);
  v16h b3 = load_bt_frag(BT, K, col0 + 48, 0);

  // steady state: unconditional next-loads, then 8 WMMAs on current frags
  int kk = 0;
  for (; kk + 32 < K; kk += 32) {
    int kn = kk + 32;
    __builtin_prefetch(A + (size_t)(row0 + (threadIdx.x & 15)) * K + kn + 24, 0, 1);
    v16h na0 = load_a_frag(A, K, row0,      kn);
    v16h na1 = load_a_frag(A, K, row0 + 16, kn);
    v16h nb0 = load_bt_frag(BT, K, col0,      kn);
    v16h nb1 = load_bt_frag(BT, K, col0 + 16, kn);
    v16h nb2 = load_bt_frag(BT, K, col0 + 32, kn);
    v16h nb3 = load_bt_frag(BT, K, col0 + 48, kn);

    acc[0][0] = WMMA_F16(a0, b0, acc[0][0]);
    acc[1][0] = WMMA_F16(a1, b0, acc[1][0]);
    acc[0][1] = WMMA_F16(a0, b1, acc[0][1]);
    acc[1][1] = WMMA_F16(a1, b1, acc[1][1]);
    acc[0][2] = WMMA_F16(a0, b2, acc[0][2]);
    acc[1][2] = WMMA_F16(a1, b2, acc[1][2]);
    acc[0][3] = WMMA_F16(a0, b3, acc[0][3]);
    acc[1][3] = WMMA_F16(a1, b3, acc[1][3]);

    a0 = na0; a1 = na1;
    b0 = nb0; b1 = nb1; b2 = nb2; b3 = nb3;
  }
  // peeled final K-step (no loads)
  acc[0][0] = WMMA_F16(a0, b0, acc[0][0]);
  acc[1][0] = WMMA_F16(a1, b0, acc[1][0]);
  acc[0][1] = WMMA_F16(a0, b1, acc[0][1]);
  acc[1][1] = WMMA_F16(a1, b1, acc[1][1]);
  acc[0][2] = WMMA_F16(a0, b2, acc[0][2]);
  acc[1][2] = WMMA_F16(a1, b2, acc[1][2]);
  acc[0][3] = WMMA_F16(a0, b3, acc[0][3]);
  acc[1][3] = WMMA_F16(a1, b3, acc[1][3]);

  int lane = threadIdx.x & 31;
  int hl   = lane >> 4;
  int nidx = lane & 15;
#pragma unroll
  for (int mi = 0; mi < 2; ++mi) {
#pragma unroll
    for (int t = 0; t < 4; ++t) {
#pragma unroll
      for (int j = 0; j < 8; ++j) {
        int rowg = row0 + mi * 16 + j + hl * 8;
        int colg = col0 + t * 16 + nidx;
        float val = acc[mi][t][j] + bias[colg];
        if (mode == MODE_QKV) {
          int p   = colg / Edim;
          int rem = colg - p * Edim;
          int hh  = rem >> 6;           // head
          int d   = rem & 63;
          int n   = rowg >> 10;         // batch
          int s   = rowg & 1023;        // seq
          size_t hb = (size_t)(n * Hdim + hh);
          if (p == 0)      qb [(hb * Sdim + s) * Ddim + d] = (f16)val;
          else if (p == 1) kb [(hb * Sdim + s) * Ddim + d] = (f16)val;
          else             vtb[(hb * Ddim + d) * Sdim + s] = (f16)val;
        } else if (mode == MODE_RES) {
          size_t idx = (size_t)rowg * Nc + colg;
          out_f32[idx] = res_in[idx] + val;
        } else { // MODE_GELU -> f16 activation buffer
          float ge = 0.5f * val * (1.0f + erff(val * 0.70710678118654752f));
          out_f16[(size_t)rowg * Nc + colg] = (f16)ge;
        }
      }
    }
  }
}

// ---------------------------------------------------------------------------
// Causal attention, one wave (32 thr) per (n, h, 16-query tile).
// Scores for all visible keys staged in 64KB LDS (unnormalized exp),
// both GEMMs via WMMA f16. Output scaled by 1/rowsum at the end.
// ---------------------------------------------------------------------------
__global__ void __launch_bounds__(32, 1)
attention_kernel(const f16* __restrict__ q,
                 const f16* __restrict__ k,
                 const f16* __restrict__ vt,
                 f16* __restrict__ attn_out) {
  __shared__ float sc[16][Sdim];   // 64 KB of the WGP's 320 KB LDS

  int qt = blockIdx.x;             // query tile (16 rows)
  int hh = blockIdx.y;
  int n  = blockIdx.z;
  int lane = threadIdx.x & 31;
  int hl   = lane >> 4;
  int nidx = lane & 15;

  const f16* qh = q  + ((size_t)(n * Hdim + hh) * Sdim) * Ddim;
  const f16* kh = k  + ((size_t)(n * Hdim + hh) * Sdim) * Ddim;
  const f16* vh = vt + ((size_t)(n * Hdim + hh) * Ddim) * Sdim;

  // Q fragments for the whole head_dim (K = 64 -> two 16x32 fragments)
  v16h a0 = load_a_frag(qh, Ddim, qt * 16, 0);
  v16h a1 = load_a_frag(qh, Ddim, qt * 16, 32);

  // ---- scores = Q K^T * 1/sqrt(D), causal masked, into LDS ----
  for (int t = 0; t <= qt; ++t) {
    v16h b0 = load_bt_frag(kh, Ddim, t * 16, 0);
    v16h b1 = load_bt_frag(kh, Ddim, t * 16, 32);
    v8f s = {};
    s = WMMA_F16(a0, b0, s);
    s = WMMA_F16(a1, b1, s);
    int key = t * 16 + nidx;
#pragma unroll
    for (int j = 0; j < 8; ++j) {
      int r = j + hl * 8;
      float val = s[j] * 0.125f;                 // 1/sqrt(64)
      if (key > qt * 16 + r) val = -1e30f;       // causal (only diag tile hits)
      sc[r][key] = val;
    }
  }
  __syncthreads();

  // ---- per-row softmax numerators in LDS, denominator in registers ----
  int L  = (qt + 1) * 16;
  int Lp = (L + 31) & ~31;                       // pad to K=32 chunks
  float dreg = 0.f;
  if (lane < 16) {
    int r = lane;
    float mx = -1e30f;
    for (int i = 0; i < L; ++i) mx = fmaxf(mx, sc[r][i]);
    float sum = 0.f;
    for (int i = 0; i < L; ++i) {
      float e = __expf(sc[r][i] - mx);
      sc[r][i] = e;
      sum += e;
    }
    for (int i = L; i < Lp; ++i) sc[r][i] = 0.f; // zero-pad partial chunk
    dreg = sum;
  }
  __syncthreads();

  // ---- out = P V  (K chunks of 32 keys), P fragments rebuilt from LDS ----
  v8f o[4] = {};
  int chunks = Lp >> 5;
  for (int c = 0; c < chunks; ++c) {
    v16h p;
#pragma unroll
    for (int e = 0; e < 16; ++e) {
      int kloc = (c << 5) + (e >> 3) * 16 + hl * 8 + (e & 7);
      p[e] = (f16)sc[nidx][kloc];
    }
#pragma unroll
    for (int t = 0; t < 4; ++t) {
      v16h b = load_bt_frag(vh, Sdim, t * 16, c << 5);
      o[t] = WMMA_F16(p, b, o[t]);
    }
  }

  // ---- scale by 1/rowsum and write merged-head layout [n, s, h*D + d] ----
#pragma unroll
  for (int t = 0; t < 4; ++t) {
#pragma unroll
    for (int j = 0; j < 8; ++j) {
      int r = j + hl * 8;
      float dn = __shfl(dreg, r, 32);
      int s = qt * 16 + r;
      int d = t * 16 + nidx;
      attn_out[((size_t)(n * Sdim + s)) * Edim + hh * Ddim + d] =
          (f16)(o[t][j] / dn);
    }
  }
}

// ---------------------------------------------------------------------------
// Host orchestration
// ---------------------------------------------------------------------------
extern "C" void kernel_launch(void* const* d_in, const int* in_sizes, int n_in,
                              void* d_out, int out_size, void* d_ws, size_t ws_size,
                              hipStream_t stream) {
  (void)in_sizes; (void)n_in; (void)out_size; (void)ws_size;

  const float* x      = (const float*)d_in[0];
  const float* ln1_g  = (const float*)d_in[1];
  const float* ln1_b  = (const float*)d_in[2];
  const float* ln2_g  = (const float*)d_in[3];
  const float* ln2_b  = (const float*)d_in[4];
  const float* w_attn = (const float*)d_in[5];
  const float* b_attn = (const float*)d_in[6];
  const float* w_proj = (const float*)d_in[7];
  const float* b_proj = (const float*)d_in[8];
  const float* w_fc   = (const float*)d_in[9];
  const float* b_fc   = (const float*)d_in[10];
  const float* w_fc2  = (const float*)d_in[11];
  const float* b_fc2  = (const float*)d_in[12];
  float* out = (float*)d_out;

  // --- workspace carve-up (256B aligned) ---
  char*  base = (char*)d_ws;
  size_t off  = 0;
  auto alloc = [&](size_t bytes) -> void* {
    void* p = base + off;
    off = (off + bytes + 255) & ~(size_t)255;
    return p;
  };
  f16* wattnT = (f16*)alloc((size_t)3 * Edim * Edim * 2);       // [2304,768]
  f16* wprojT = (f16*)alloc((size_t)Edim * Edim * 2);           // [768,768]
  f16* wfcT   = (f16*)alloc((size_t)FFdim * Edim * 2);          // [3072,768]
  f16* wfc2T  = (f16*)alloc((size_t)Edim * FFdim * 2);          // [768,3072]
  float* x1   = (float*)alloc((size_t)Mdim * Edim * 4);         // residual 1
  f16* hbuf   = (f16*)alloc((size_t)Mdim * Edim * 2);           // ln output (reused)
  f16* qb     = (f16*)alloc((size_t)Ndim * Hdim * Sdim * Ddim * 2);
  f16* kb     = (f16*)alloc((size_t)Ndim * Hdim * Sdim * Ddim * 2);
  f16* vtb    = (f16*)alloc((size_t)Ndim * Hdim * Ddim * Sdim * 2);
  f16* aob    = (f16*)alloc((size_t)Mdim * Edim * 2);           // attn out (f16)
  f16* ffb    = qb;  // alias: q/k/vT/attn_out dead once proj GEMM done;
                     // region is exactly Mdim*FFdim*2 bytes (50.3 MB)

  auto cvt = [&](const float* src, f16* dst, int K, int N) {
    size_t total = (size_t)K * N;
    convert_transpose_kernel<<<dim3((unsigned)((total + 255) / 256)), 256, 0, stream>>>(
        src, dst, K, N);
  };
  cvt(w_attn, wattnT, Edim, 3 * Edim);
  cvt(w_proj, wprojT, Edim, Edim);
  cvt(w_fc,   wfcT,   Edim, FFdim);
  cvt(w_fc2,  wfc2T,  FFdim, Edim);

  // ---- attention branch ----
  layernorm_f16_kernel<<<Mdim, 256, 0, stream>>>(x, ln1_g, ln1_b, hbuf);

  gemm_wmma_kernel<<<dim3((3 * Edim) / 256, Mdim / 64), 256, 0, stream>>>(
      hbuf, wattnT, b_attn, Mdim, Edim, 3 * Edim, MODE_QKV,
      nullptr, nullptr, nullptr, qb, kb, vtb);

  attention_kernel<<<dim3(Sdim / 16, Hdim, Ndim), 32, 0, stream>>>(qb, kb, vtb, aob);

  gemm_wmma_kernel<<<dim3(Edim / 256, Mdim / 64), 256, 0, stream>>>(
      aob, wprojT, b_proj, Mdim, Edim, Edim, MODE_RES,
      x1, x, nullptr, nullptr, nullptr, nullptr);

  // ---- MLP branch ----
  layernorm_f16_kernel<<<Mdim, 256, 0, stream>>>(x1, ln2_g, ln2_b, hbuf);

  gemm_wmma_kernel<<<dim3(FFdim / 256, Mdim / 64), 256, 0, stream>>>(
      hbuf, wfcT, b_fc, Mdim, Edim, FFdim, MODE_GELU,
      nullptr, nullptr, ffb, nullptr, nullptr, nullptr);

  gemm_wmma_kernel<<<dim3(Edim / 256, Mdim / 64), 256, 0, stream>>>(
      ffb, wfc2T, b_fc2, Mdim, FFdim, Edim, MODE_RES,
      out, x1, nullptr, nullptr, nullptr, nullptr);
}
